// MultiHeadGAT_51041391346030
// MI455X (gfx1250) — matrix operand order repaired and checked by reading, not compiled, not run on hardware
//
#include <hip/hip_runtime.h>
#include <math.h>

// ---------------- problem constants (from reference) ----------------
#define HEADS   4
#define HID     64
#define HC      256     // HEADS*HID
#define IN_DIM  128
#define NEG_SLOPE 0.2f
#define NUM_ACTIONS 10

typedef __attribute__((ext_vector_type(16))) _Float16 v16h;
typedef __attribute__((ext_vector_type(8)))  _Float16 v8h;
typedef __attribute__((ext_vector_type(8)))  float    v8f;

// Order-preserving float<->uint key for atomicMax-based segment max.
__device__ __forceinline__ unsigned f32_key(float f) {
    unsigned b = __float_as_uint(f);
    return (b & 0x80000000u) ? ~b : (b | 0x80000000u);
}
__device__ __forceinline__ float key_f32(unsigned k) {
    return __uint_as_float((k & 0x80000000u) ? (k & 0x7FFFFFFFu) : ~k);
}
#define AMAX_INIT 0x007FFFFFu   // f32_key(-inf)

// ---------------- utility kernels ----------------
__global__ void k_zero_f32(float* p, long n) {
    long i = (long)blockIdx.x * blockDim.x + threadIdx.x;
    if (i < n) p[i] = 0.f;
}
__global__ void k_fill_u32(unsigned* p, long n, unsigned v) {
    long i = (long)blockIdx.x * blockDim.x + threadIdx.x;
    if (i < n) p[i] = v;
}
__global__ void k_cvt_f16(const float* __restrict__ s, _Float16* __restrict__ d, long n) {
    long i = (long)blockIdx.x * blockDim.x + threadIdx.x;
    if (i < n) d[i] = (_Float16)s[i];
}

// ---- weight pack: f32 row-major [K x cols] -> f16 WMMA-fragment order ----
// Layout consumed by the GEMM: for tile nt, k-block kb, lane L:
//   Bp[((nt*KB + kb) << 9) + (L << 4) + j] = B[(kb*32 + (L>>4)*16 + j) * cols + nt*16 + (L&15)]
// (zero-padded for columns >= cols). One thread per (nt, kb, lane).
__global__ void k_pack_w(const float* __restrict__ B, _Float16* __restrict__ Bp,
                         int cols, int Nld, int K) {
    int i = blockIdx.x * blockDim.x + threadIdx.x;
    const int Ntiles = Nld >> 4, KB = K >> 5;
    if (i >= Ntiles * KB * 32) return;
    int lane = i & 31;
    int t = i >> 5;
    int kb = t % KB, nt = t / KB;
    int hi = lane >> 4, ml = lane & 15;
    int n = (nt << 4) + ml;
    int kbase = (kb << 5) + (hi << 4);
    _Float16* dst = Bp + ((size_t)(nt * KB + kb) << 9) + (lane << 4);
    #pragma unroll
    for (int j = 0; j < 16; ++j)
        dst[j] = (n < cols) ? (_Float16)B[(size_t)(kbase + j) * cols + n] : (_Float16)0.f;
}

// ---------------- degree + mean self-loop attr ----------------
__global__ void k_deg_loop(const int* __restrict__ ei, const float* __restrict__ ea,
                           float* deg, float* loopattr, int E) {
    int e = blockIdx.x * blockDim.x + threadIdx.x;
    if (e >= E) return;
    int c = ei[E + e];
    atomicAdd(&deg[c], 1.f);
    atomicAdd(&loopattr[2 * (size_t)c],     ea[2 * (size_t)e]);
    atomicAdd(&loopattr[2 * (size_t)c + 1], ea[2 * (size_t)e + 1]);
}
__global__ void k_loop_div(float* loopattr, const float* deg, int N) {
    int i = blockIdx.x * blockDim.x + threadIdx.x;
    if (i >= N) return;
    float d = fmaxf(deg[i], 1.f);
    loopattr[2 * (size_t)i]     /= d;
    loopattr[2 * (size_t)i + 1] /= d;
}

// ---------------- WMMA GEMM: C[M,Nld] = A[M,K] @ B[K,Nld] ----------------
// A row-major f16 (lda=K); B pre-packed to fragment order by k_pack_w; C f32 row-major.
// Each wave owns 2 M-tiles x NT N-tiles: every B fragment fetched once feeds 2 WMMAs,
// every A fragment feeds NT WMMAs (2*NT wmma per (2+NT) b128-pair loads).
template<int NT>
__global__ void k_wmma_gemm(const _Float16* __restrict__ A, const _Float16* __restrict__ Bp,
                            float* __restrict__ C, int M, int Nld, int K) {
    const int lane = threadIdx.x;            // 0..31
    const int Ntiles = Nld >> 4;
    const int KB = K >> 5;
    const int nt0 = (blockIdx.y * blockDim.y + threadIdx.y) * NT;
    if (nt0 >= Ntiles) return;
    const int m0 = blockIdx.x << 5;          // two 16-row stripes per block
    const int hi = lane >> 4;                // lane half selector
    const int ml = lane & 15;

    int ar0 = m0 + ml;      if (ar0 >= M) ar0 = M - 1;   // clamp; padded rows discarded
    int ar1 = m0 + 16 + ml; if (ar1 >= M) ar1 = M - 1;
    const _Float16* Ap0 = A + (size_t)ar0 * K + (hi << 4);    // +0 or +16
    const _Float16* Ap1 = A + (size_t)ar1 * K + (hi << 4);

    v8f acc0[NT] = {};
    v8f acc1[NT] = {};
    for (int kb = 0; kb < KB; ++kb) {
        // A: lanes 0-15 hold K = kb*32+{0..15}; lanes 16-31 K = kb*32+{16..31}
        v8h alo0 = *(const v8h*)(Ap0 + (kb << 5));
        v8h ahi0 = *(const v8h*)(Ap0 + (kb << 5) + 8);
        v8h alo1 = *(const v8h*)(Ap1 + (kb << 5));
        v8h ahi1 = *(const v8h*)(Ap1 + (kb << 5) + 8);
        v16h a0, a1;
        #pragma unroll
        for (int i = 0; i < 8; ++i) {
            a0[i] = alo0[i]; a0[i + 8] = ahi0[i];
            a1[i] = alo1[i]; a1[i + 8] = ahi1[i];
        }
        #pragma unroll
        for (int t = 0; t < NT; ++t) {
            // one 32B vector load per B fragment (packed order), reused by both M tiles
            const v16h b = *(const v16h*)(Bp + ((size_t)((nt0 + t) * KB + kb) << 9)
                                             + (lane << 4));
            acc0[t] = __builtin_amdgcn_wmma_f32_16x16x32_f16(false, a0, false, b,
                                                             (short)0, acc0[t], false, false);
            acc1[t] = __builtin_amdgcn_wmma_f32_16x16x32_f16(false, a1, false, b,
                                                             (short)0, acc1[t], false, false);
        }
    }
    // C: element r -> row stripe_base + r + 8*hi, col nt*16 + ml
    #pragma unroll
    for (int mt = 0; mt < 2; ++mt) {
        const int s0 = m0 + (mt << 4);
        const int rbase = s0 + (hi << 3);
        if (s0 + 16 <= M) {                   // straight-line fast path
            #pragma unroll
            for (int t = 0; t < NT; ++t) {
                float* cp = C + (size_t)rbase * Nld + ((nt0 + t) << 4) + ml;
                const v8f* av = mt ? acc1 : acc0;
                #pragma unroll
                for (int r = 0; r < 8; ++r) cp[(size_t)r * Nld] = av[t][r];
            }
        } else if (s0 < M) {
            #pragma unroll
            for (int t = 0; t < NT; ++t) {
                float* cp = C + (size_t)rbase * Nld + ((nt0 + t) << 4) + ml;
                const v8f* av = mt ? acc1 : acc0;
                #pragma unroll
                for (int r = 0; r < 8; ++r)
                    if (rbase + r < M) cp[(size_t)r * Nld] = av[t][r];
            }
        }
    }
}

// ---------------- attention scalars ----------------
__global__ void k_alpha_nodes(const float* __restrict__ h, const float* __restrict__ av_s,
                              const float* __restrict__ av_d,
                              float* asrc, float* adst, int N) {
    int i = blockIdx.x * blockDim.x + threadIdx.x;
    if (i >= N * HEADS) return;
    int n = i >> 2, hh = i & 3;
    const float* hp = h + (size_t)n * HC + hh * HID;
    const float* sa = av_s + hh * HID;
    const float* da = av_d + hh * HID;
    float s = 0.f, d = 0.f;
    #pragma unroll 8
    for (int c = 0; c < HID; ++c) { float v = hp[c]; s += v * sa[c]; d += v * da[c]; }
    asrc[i] = s; adst[i] = d;
}

// proj[row*4+h] = sum_c We[row, h*HID+c] * att_edge[h,c]
__global__ void k_eproj(const float* __restrict__ We, const float* __restrict__ ae,
                        float* proj) {
    int t = threadIdx.x;
    if (t >= 8) return;
    int row = t >> 2, hh = t & 3;
    float s = 0.f;
    for (int c = 0; c < HID; ++c) s += We[row * HC + hh * HID + c] * ae[hh * HID + c];
    proj[t] = s;
}

// per-edge leaky-relu alpha + segment max (edges e<E real, e>=E virtual self-loop)
__global__ void k_edge_alpha(const int* __restrict__ ei, const float* __restrict__ ea,
                             const float* __restrict__ loopattr,
                             const float* __restrict__ asrc, const float* __restrict__ adst,
                             const float* __restrict__ proj,
                             float* alpha, unsigned* amaxk, int E, int N) {
    int e = blockIdx.x * blockDim.x + threadIdx.x;
    if (e >= E + N) return;
    int r, c; float e0, e1;
    if (e < E) {
        r = ei[e]; c = ei[E + e];
        e0 = ea[2 * (size_t)e]; e1 = ea[2 * (size_t)e + 1];
    } else {
        r = c = e - E;
        e0 = loopattr[2 * (size_t)r]; e1 = loopattr[2 * (size_t)r + 1];
    }
    #pragma unroll
    for (int hh = 0; hh < HEADS; ++hh) {
        float al = asrc[r * 4 + hh] + adst[c * 4 + hh] + e0 * proj[hh] + e1 * proj[4 + hh];
        al = al > 0.f ? al : NEG_SLOPE * al;
        alpha[(size_t)e * 4 + hh] = al;
        atomicMax(&amaxk[c * 4 + hh], f32_key(al));
    }
}

// exp(alpha - amax) in place + segment sum
__global__ void k_edge_exp(const int* __restrict__ ei, const unsigned* __restrict__ amaxk,
                           float* alpha, float* denom, int E, int N) {
    long i = (long)blockIdx.x * blockDim.x + threadIdx.x;
    if (i >= (long)(E + N) * 4) return;
    int e = (int)(i >> 2), hh = (int)(i & 3);
    int c = (e < E) ? ei[E + e] : e - E;
    unsigned k = amaxk[c * 4 + hh];
    float m = (k == AMAX_INIT) ? 0.f : key_f32(k);   // mirror isfinite guard
    float ex = __expf(alpha[i] - m);
    alpha[i] = ex;
    atomicAdd(&denom[c * 4 + hh], ex);
}

// one 256-thread block per edge: agg[col, :] += h[row, :] * att
__global__ void k_scatter(const int* __restrict__ ei, const float* __restrict__ h,
                          const float* __restrict__ alpha, const float* __restrict__ denom,
                          float* agg, int E, int N) {
    int e = blockIdx.x;
    int t = threadIdx.x;          // 0..255
    int hh = t >> 6;
    int r, c;
    if (e < E) { r = ei[e]; c = ei[E + e]; } else { r = c = e - E; }
    float att = alpha[(size_t)e * 4 + hh] / (denom[c * 4 + hh] + 1e-16f);
    atomicAdd(&agg[(size_t)c * HC + t], h[(size_t)r * HC + t] * att);
}

// ---------------- epilogues ----------------
__global__ void k_bias_elu(const float* __restrict__ agg, const float* __restrict__ b,
                           float* out, long n) {
    long i = (long)blockIdx.x * blockDim.x + threadIdx.x;
    if (i >= n) return;
    int j = (int)(i & (HC - 1));
    float x = agg[i] + b[j];
    out[i] = x > 0.f ? x : __expf(x) - 1.f;
}
__global__ void k_mean_bias_elu(const float* __restrict__ agg, const float* __restrict__ b,
                                float* out, int N) {
    int i = blockIdx.x * blockDim.x + threadIdx.x;
    if (i >= N * HID) return;
    int n = i >> 6, c = i & 63;
    const float* p = agg + (size_t)n * HC + c;
    float x = 0.25f * (p[0] + p[64] + p[128] + p[192]) + b[c];
    out[i] = x > 0.f ? x : __expf(x) - 1.f;
}

// final clipped concat: wire [N,nw] | terminal [N,nt] | action [N,10]
__global__ void k_gather_out(const float* __restrict__ wl, const float* __restrict__ tl,
                             const float* __restrict__ al,
                             const float* __restrict__ bw, const float* __restrict__ bt,
                             const float* __restrict__ ba,
                             const int* pnw, const int* pnt, float* out, int N) {
    int n = blockIdx.x;
    int t = threadIdx.x;          // 0..127
    int nw = *pnw, nt_ = *pnt;
    if (t < nw) {
        out[(size_t)n * nw + t] = wl[(size_t)n * 64 + t] + bw[t];
    } else if (t < nw + nt_) {
        int j = t - nw;
        out[(size_t)N * nw + (size_t)n * nt_ + j] = tl[(size_t)n * 32 + j] + bt[j];
    } else if (t < nw + nt_ + NUM_ACTIONS) {
        int j = t - nw - nt_;
        out[(size_t)N * nw + (size_t)N * nt_ + (size_t)n * NUM_ACTIONS + j]
            = al[(size_t)n * 16 + j] + ba[j];
    }
}

// ---------------- host orchestration ----------------
#define L1D(kern, cnt, ...) \
    kern<<<dim3((unsigned)((((long)(cnt)) + 255) / 256)), dim3(256), 0, stream>>>(__VA_ARGS__)

extern "C" void kernel_launch(void* const* d_in, const int* in_sizes, int n_in,
                              void* d_out, int out_size, void* d_ws, size_t ws_size,
                              hipStream_t stream) {
    const float* x   = (const float*)d_in[0];
    const float* ea  = (const float*)d_in[1];
    const float* W1  = (const float*)d_in[2];
    const float* as1 = (const float*)d_in[3];
    const float* ad1 = (const float*)d_in[4];
    const float* We1 = (const float*)d_in[5];
    const float* ae1 = (const float*)d_in[6];
    const float* b1  = (const float*)d_in[7];
    const float* W2  = (const float*)d_in[8];
    const float* as2 = (const float*)d_in[9];
    const float* ad2 = (const float*)d_in[10];
    const float* We2 = (const float*)d_in[11];
    const float* ae2 = (const float*)d_in[12];
    const float* b2  = (const float*)d_in[13];
    const float* Ww  = (const float*)d_in[14];
    const float* bw  = (const float*)d_in[15];
    const float* Wt  = (const float*)d_in[16];
    const float* bt  = (const float*)d_in[17];
    const float* Wa  = (const float*)d_in[18];
    const float* ba  = (const float*)d_in[19];
    const int*   ei  = (const int*)d_in[20];
    const int*   pnw = (const int*)d_in[21];
    const int*   pnt = (const int*)d_in[22];
    float* out = (float*)d_out;

    const int N  = in_sizes[0] / IN_DIM;
    const int E  = in_sizes[20] / 2;
    const int Ef = E + N;

    // ---- workspace carve (256B aligned) ----
    char* base = (char*)d_ws; size_t off = 0;
    auto carve = [&](size_t bytes) -> void* {
        void* p = base + off;
        off = (off + bytes + 255) & ~(size_t)255;
        return p;
    };
    float*     deg      = (float*)carve((size_t)N * 4);
    float*     loopattr = (float*)carve((size_t)N * 2 * 4);
    _Float16*  xh       = (_Float16*)carve((size_t)N * IN_DIM * 2);
    _Float16*  W1p      = (_Float16*)carve((size_t)IN_DIM * HC * 2);   // packed
    _Float16*  W2p      = (_Float16*)carve((size_t)HC * HC * 2);       // packed
    _Float16*  Wwp      = (_Float16*)carve((size_t)HID * 64 * 2);      // packed
    _Float16*  Wtp      = (_Float16*)carve((size_t)HID * 32 * 2);      // packed
    _Float16*  Wap      = (_Float16*)carve((size_t)HID * 16 * 2);      // packed (10->16)
    float*     hbuf     = (float*)carve((size_t)N * HC * 4);     // lin out, both layers
    float*     asrc     = (float*)carve((size_t)N * HEADS * 4);
    float*     adst     = (float*)carve((size_t)N * HEADS * 4);
    float*     eproj    = (float*)carve(8 * 4);
    float*     alpha    = (float*)carve((size_t)Ef * HEADS * 4);
    unsigned*  amaxk    = (unsigned*)carve((size_t)N * HEADS * 4);
    float*     denom    = (float*)carve((size_t)N * HEADS * 4);
    float*     agg      = (float*)carve((size_t)N * HC * 4);     // reused both layers
    float*     h1act    = (float*)carve((size_t)N * HC * 4);
    _Float16*  h1h      = (_Float16*)carve((size_t)N * HC * 2);
    float*     h2       = (float*)carve((size_t)N * HID * 4);
    _Float16*  h2h      = (_Float16*)carve((size_t)N * HID * 2);
    float*     wl       = (float*)carve((size_t)N * 64 * 4);
    float*     tl       = (float*)carve((size_t)N * 32 * 4);
    float*     al       = (float*)carve((size_t)N * 16 * 4);
    (void)ws_size; (void)n_in; (void)out_size;

    auto packw = [&](const float* W, _Float16* Wp, int cols, int Nld, int K) {
        int cnt = (Nld / 16) * (K / 32) * 32;
        k_pack_w<<<dim3((unsigned)((cnt + 255) / 256)), dim3(256), 0, stream>>>(
            W, Wp, cols, Nld, K);
    };
    auto gemm = [&](const _Float16* A, const _Float16* Bp, float* C, int M, int Nld, int K) {
        unsigned Mt = (unsigned)((M + 31) / 32);        // 2 M tiles per block
        int Ntiles = Nld / 16;
        if (Ntiles % 16 == 0) {
            k_wmma_gemm<4><<<dim3(Mt, (unsigned)(Ntiles / 16)), dim3(32, 4), 0, stream>>>(
                A, Bp, C, M, Nld, K);
        } else if (Ntiles % 4 == 0) {
            k_wmma_gemm<4><<<dim3(Mt, (unsigned)(Ntiles / 4)), dim3(32, 1), 0, stream>>>(
                A, Bp, C, M, Nld, K);
        } else if (Ntiles % 2 == 0) {
            k_wmma_gemm<2><<<dim3(Mt, (unsigned)(Ntiles / 2)), dim3(32, 1), 0, stream>>>(
                A, Bp, C, M, Nld, K);
        } else {
            k_wmma_gemm<1><<<dim3(Mt, (unsigned)Ntiles), dim3(32, 1), 0, stream>>>(
                A, Bp, C, M, Nld, K);
        }
    };

    // ---- precompute: f16 A operand + packed weights + self-loop mean attrs ----
    L1D(k_cvt_f16, (long)N * IN_DIM, x, xh, (long)N * IN_DIM);
    packw(W1, W1p, HC, HC, IN_DIM);
    packw(W2, W2p, HC, HC, HC);
    packw(Ww, Wwp, 64, 64, HID);
    packw(Wt, Wtp, 32, 32, HID);
    packw(Wa, Wap, NUM_ACTIONS, 16, HID);
    L1D(k_zero_f32, N, deg, (long)N);
    L1D(k_zero_f32, 2 * N, loopattr, (long)2 * N);
    L1D(k_deg_loop, E, ei, ea, deg, loopattr, E);
    L1D(k_loop_div, N, loopattr, deg, N);

    // ---- layer 1 (concat) ----
    gemm(xh, W1p, hbuf, N, HC, IN_DIM);
    L1D(k_alpha_nodes, N * HEADS, hbuf, as1, ad1, asrc, adst, N);
    k_eproj<<<dim3(1), dim3(32), 0, stream>>>(We1, ae1, eproj);
    L1D(k_fill_u32, N * HEADS, amaxk, (long)N * HEADS, AMAX_INIT);
    L1D(k_zero_f32, N * HEADS, denom, (long)N * HEADS);
    L1D(k_zero_f32, (long)N * HC, agg, (long)N * HC);
    L1D(k_edge_alpha, Ef, ei, ea, loopattr, asrc, adst, eproj, alpha, amaxk, E, N);
    L1D(k_edge_exp, (long)Ef * 4, ei, amaxk, alpha, denom, E, N);
    k_scatter<<<dim3((unsigned)Ef), dim3(256), 0, stream>>>(ei, hbuf, alpha, denom, agg, E, N);
    L1D(k_bias_elu, (long)N * HC, agg, b1, h1act, (long)N * HC);
    L1D(k_cvt_f16, (long)N * HC, h1act, h1h, (long)N * HC);

    // ---- layer 2 (mean over heads) ----
    gemm(h1h, W2p, hbuf, N, HC, HC);
    L1D(k_alpha_nodes, N * HEADS, hbuf, as2, ad2, asrc, adst, N);
    k_eproj<<<dim3(1), dim3(32), 0, stream>>>(We2, ae2, eproj);
    L1D(k_fill_u32, N * HEADS, amaxk, (long)N * HEADS, AMAX_INIT);
    L1D(k_zero_f32, N * HEADS, denom, (long)N * HEADS);
    L1D(k_zero_f32, (long)N * HC, agg, (long)N * HC);
    L1D(k_edge_alpha, Ef, ei, ea, loopattr, asrc, adst, eproj, alpha, amaxk, E, N);
    L1D(k_edge_exp, (long)Ef * 4, ei, amaxk, alpha, denom, E, N);
    k_scatter<<<dim3((unsigned)Ef), dim3(256), 0, stream>>>(ei, hbuf, alpha, denom, agg, E, N);
    L1D(k_mean_bias_elu, N * HID, agg, b2, h2, N);
    L1D(k_cvt_f16, (long)N * HID, h2, h2h, (long)N * HID);

    // ---- output heads (WMMA into padded scratch, then clipped concat) ----
    gemm(h2h, Wwp, wl, N, 64, HID);
    gemm(h2h, Wtp, tl, N, 32, HID);
    gemm(h2h, Wap, al, N, 16, HID);
    k_gather_out<<<dim3((unsigned)N), dim3(128), 0, stream>>>(wl, tl, al, bw, bt, ba,
                                                              pnw, pnt, out, N);
}
